// Model_11450382811541
// MI455X (gfx1250) — matrix operand order repaired and checked by reading, not compiled, not run on hardware
//
#include <hip/hip_runtime.h>
#include <hip/hip_bf16.h>

// ---------------- types for WMMA ----------------
typedef __attribute__((ext_vector_type(16))) __bf16 v16bf;
typedef __attribute__((ext_vector_type(8)))  __bf16 v8bf;
typedef __attribute__((ext_vector_type(8)))  float  v8f;

// matches the async-load builtin's parameter pointee type exactly
typedef int v4i_g __attribute__((vector_size(16)));

#define AS1 __attribute__((address_space(1)))
#define AS3 __attribute__((address_space(3)))

// Async global->LDS path (gfx1250). Guarded: falls back to the proven
// synchronous load+ds_store path if the builtin is unavailable.
#if defined(__HIP_DEVICE_COMPILE__) && defined(__gfx1250__) && \
    __has_builtin(__builtin_amdgcn_global_load_async_to_lds_b128)
#define USE_ASYNC 1
#else
#define USE_ASYNC 0
#endif

// ---------------- problem sizes ----------------
#define BATCH  4096
#define H_IN   62
#define W_IN   47
#define HO     60
#define WO     45
#define HW     (HO*WO)        // 2700
#define KFEAT  32400          // 12 * 2700
#define KP     32416          // KFEAT padded to multiple of 32 (32*1013)
#define NCLS   1000
#define NP     1024           // N padded for tiling

// GEMM tiling
#define BM 128
#define BN 128
#define BK 32
#define NK (KP/BK)            // 1013
#define LDSS 40               // padded LDS row stride (bf16 elems) -> 80B, conflict-free

// ---------------------------------------------------------------------------
// conv1: (B,1,62,47) -> (B,3,60,45), k=5, pad=1, valid crop 55x40, leaky 0.01
// ---------------------------------------------------------------------------
__global__ __launch_bounds__(256) void conv1_kernel(
    const float* __restrict__ x, const float* __restrict__ K1,
    float* __restrict__ out)
{
    int idx = blockIdx.x * 256 + threadIdx.x;             // over 4096*3*2700
    int xw = idx % WO;
    int t  = idx / WO;
    int yh = t % HO;  t /= HO;
    int o  = t % 3;
    int b  = t / 3;
    float v = 0.0f;
    if (yh < 55 && xw < 40) {
        const float* xin = x + (size_t)b * (H_IN * W_IN);
        #pragma unroll
        for (int dy = 0; dy < 5; ++dy) {
            int iy = yh + dy - 1;
            if ((unsigned)iy >= (unsigned)H_IN) continue;
            #pragma unroll
            for (int dx = 0; dx < 5; ++dx) {
                int ix = xw + dx - 1;
                if ((unsigned)ix >= (unsigned)W_IN) continue;
                // quirky reshape: w(dy,dx,c=0,o) = K1[(dy*5+dx)*3 + o]
                v += xin[iy * W_IN + ix] * K1[(dy * 5 + dx) * 3 + o];
            }
        }
        v = v > 0.0f ? v : 0.01f * v;                     // leaky relu
    }
    out[idx] = v;
}

// ---------------------------------------------------------------------------
// conv2: (B,3,60,45) -> (B,6,60,45), k=3, pad=1, valid crop 57x42, leaky 0.01
// ---------------------------------------------------------------------------
__global__ __launch_bounds__(256) void conv2_kernel(
    const float* __restrict__ in, const float* __restrict__ K2,
    float* __restrict__ out)
{
    int idx = blockIdx.x * 256 + threadIdx.x;             // over 4096*6*2700
    int xw = idx % WO;
    int t  = idx / WO;
    int yh = t % HO;  t /= HO;
    int o  = t % 6;
    int b  = t / 6;
    float v = 0.0f;
    if (yh < 57 && xw < 42) {
        #pragma unroll
        for (int c = 0; c < 3; ++c) {
            const float* cin = in + ((size_t)b * 3 + c) * HW;
            #pragma unroll
            for (int dy = 0; dy < 3; ++dy) {
                int iy = yh + dy - 1;
                if ((unsigned)iy >= (unsigned)HO) continue;
                #pragma unroll
                for (int dx = 0; dx < 3; ++dx) {
                    int ix = xw + dx - 1;
                    if ((unsigned)ix >= (unsigned)WO) continue;
                    v += cin[iy * WO + ix] * K2[(c * 9 + dy * 3 + dx) * 6 + o];
                }
            }
        }
        v = v > 0.0f ? v : 0.01f * v;
    }
    out[idx] = v;
}

// ---------------------------------------------------------------------------
// conv3: (B,6,60,45) -> (B,12,60,45), k=3, pad=1, valid crop 57x42, NO relu.
// Writes bf16 directly into FC activation matrix A[b][o*2700+y*45+x].
// ---------------------------------------------------------------------------
__global__ __launch_bounds__(256) void conv3_kernel(
    const float* __restrict__ in, const float* __restrict__ K3,
    __bf16* __restrict__ Abf)
{
    int idx = blockIdx.x * 256 + threadIdx.x;             // over 4096*12*2700
    int xw = idx % WO;
    int t  = idx / WO;
    int yh = t % HO;  t /= HO;
    int o  = t % 12;
    int b  = t / 12;
    float v = 0.0f;
    if (yh < 57 && xw < 42) {
        #pragma unroll
        for (int c = 0; c < 6; ++c) {
            const float* cin = in + ((size_t)b * 6 + c) * HW;
            #pragma unroll
            for (int dy = 0; dy < 3; ++dy) {
                int iy = yh + dy - 1;
                if ((unsigned)iy >= (unsigned)HO) continue;
                #pragma unroll
                for (int dx = 0; dx < 3; ++dx) {
                    int ix = xw + dx - 1;
                    if ((unsigned)ix >= (unsigned)WO) continue;
                    v += cin[iy * WO + ix] * K3[(c * 9 + dy * 3 + dx) * 12 + o];
                }
            }
        }
    }
    int col = (o * HO + yh) * WO + xw;                    // 0..32399
    Abf[(size_t)b * KP + col] = (__bf16)v;
}

// Zero the K-padding columns of A (cols 32400..32415 for every row).
__global__ __launch_bounds__(256) void apad_kernel(__bf16* __restrict__ Abf)
{
    int idx = blockIdx.x * 256 + threadIdx.x;             // over 4096*16
    int j = idx & 15;
    int b = idx >> 4;
    Abf[(size_t)b * KP + KFEAT + j] = (__bf16)0.0f;
}

// fc_w (1000 x 32400 f32) -> Wbf (1024 x 32416 bf16, zero padded)
__global__ __launch_bounds__(256) void wcvt_kernel(
    const float* __restrict__ fcw, __bf16* __restrict__ Wbf)
{
    size_t base = ((size_t)blockIdx.x * 256 + threadIdx.x) * 4;  // NP*KP total
    #pragma unroll
    for (int j = 0; j < 4; ++j) {
        size_t e = base + j;
        int k = (int)(e % KP);
        int n = (int)(e / KP);
        float v = (n < NCLS && k < KFEAT) ? fcw[(size_t)n * KFEAT + k] : 0.0f;
        Wbf[e] = (__bf16)v;
    }
}

// ---------------------------------------------------------------------------
// FC GEMM: out[4096][1000] = A[M][KP] * Wbf[N][KP]^T + bias, bf16 WMMA.
// Block = 128x128, 8 waves (2 in M x 4 in N), each wave 4x2 16x16 tiles.
// Double-buffered LDS fed by GLOBAL_LOAD_ASYNC_TO_LDS_B128 (ASYNCcnt),
// one barrier per K-step, next tile streams in during current tile's WMMAs.
// ---------------------------------------------------------------------------
__global__ __launch_bounds__(256) void fc_gemm_kernel(
    const __bf16* __restrict__ A, const __bf16* __restrict__ W,
    const float* __restrict__ bias, float* __restrict__ out)
{
    __shared__ __bf16 As[2][BM * LDSS];
    __shared__ __bf16 Bs[2][BN * LDSS];

    const int tid   = threadIdx.x;
    const int lane  = tid & 31;
    const int wave  = tid >> 5;
    const int waveM = wave >> 2;          // 0..1  -> 64 rows each
    const int waveN = wave & 3;           // 0..3  -> 32 cols each
    const int hl    = lane >> 4;          // half-wave select
    const int l15   = lane & 15;

    // cooperative load mapping: 256 thr, each 16 bf16 (32B) per matrix
    const int loadRow  = tid >> 1;        // 0..127
    const int loadHalf = tid & 1;         // 16-elem chunk

    const __bf16* gA = A + (size_t)(blockIdx.x * BM + loadRow) * KP + loadHalf * 16;
    const __bf16* gW = W + (size_t)(blockIdx.y * BN + loadRow) * KP + loadHalf * 16;

    __bf16* ldsA[2] = { &As[0][loadRow * LDSS + loadHalf * 16],
                        &As[1][loadRow * LDSS + loadHalf * 16] };
    __bf16* ldsB[2] = { &Bs[0][loadRow * LDSS + loadHalf * 16],
                        &Bs[1][loadRow * LDSS + loadHalf * 16] };

    v8f acc[4][2];
    #pragma unroll
    for (int i = 0; i < 4; ++i)
        #pragma unroll
        for (int j = 0; j < 2; ++j)
            acc[i][j] = (v8f)0.0f;

    // 8 WMMAs on one K-step tile held in LDS buffer (curA/curB).
    auto compute = [&](const __bf16* curA, const __bf16* curB) {
        // A frag: lanes 0-15 row m hold K {0..7,16..23}; lanes 16-31 {8..15,24..31}
        v16bf af[4];
        #pragma unroll
        for (int sm = 0; sm < 4; ++sm) {
            int r = waveM * 64 + sm * 16 + l15;
            v8bf lo = *(const v8bf*)&curA[r * LDSS + hl * 8];
            v8bf hi = *(const v8bf*)&curA[r * LDSS + hl * 8 + 16];
            af[sm] = __builtin_shufflevector(lo, hi,
                0,1,2,3,4,5,6,7,8,9,10,11,12,13,14,15);
        }
        // B frag: lanes 0-15 col n hold K {0..7,16..23}; lanes 16-31 {8..15,24..31}
        v16bf bf[2];
        #pragma unroll
        for (int sn = 0; sn < 2; ++sn) {
            int r = waveN * 32 + sn * 16 + l15;
            v8bf lo = *(const v8bf*)&curB[r * LDSS + hl * 8];
            v8bf hi = *(const v8bf*)&curB[r * LDSS + hl * 8 + 16];
            bf[sn] = __builtin_shufflevector(lo, hi,
                0,1,2,3,4,5,6,7,8,9,10,11,12,13,14,15);
        }
        #pragma unroll
        for (int sm = 0; sm < 4; ++sm)
            #pragma unroll
            for (int sn = 0; sn < 2; ++sn)
                acc[sm][sn] = __builtin_amdgcn_wmma_f32_16x16x32_bf16(
                    false, af[sm], false, bf[sn], (short)0, acc[sm][sn],
                    false, false);
    };

#if USE_ASYNC
    // Issue this thread's 4 async b128 transfers for K-step kt into buffer b.
    // Builtin signature (from clang diagnostic): arg0 = AS1 int4* (global src),
    // arg1 = AS3 int4* (LDS dst), then imm offset, imm cpol. The IOFFSET is
    // applied to BOTH global and LDS addresses (ISA 08, async load pseudocode).
    auto issue = [&](int kt, int b) {
        AS1 v4i_g* ga = (AS1 v4i_g*)(gA + (size_t)kt * BK);
        AS1 v4i_g* gw = (AS1 v4i_g*)(gW + (size_t)kt * BK);
        AS3 v4i_g* la = (AS3 v4i_g*)ldsA[b];
        AS3 v4i_g* lb = (AS3 v4i_g*)ldsB[b];
        __builtin_amdgcn_global_load_async_to_lds_b128(ga, la, 0,  0);
        __builtin_amdgcn_global_load_async_to_lds_b128(ga, la, 16, 0);
        __builtin_amdgcn_global_load_async_to_lds_b128(gw, lb, 0,  0);
        __builtin_amdgcn_global_load_async_to_lds_b128(gw, lb, 16, 0);
    };

    issue(0, 0);                                   // prologue: tile 0 -> buf 0
    int kt = 0;
    for (; kt + 1 < NK; ++kt) {
        asm volatile("s_wait_asynccnt 0" ::: "memory");  // own tile-kt loads landed
        __syncthreads();                           // all waves' data visible,
                                                   // prev buffer's readers done
        issue(kt + 1, (kt + 1) & 1);               // stream next tile during WMMAs
        compute(As[kt & 1], Bs[kt & 1]);
    }
    asm volatile("s_wait_asynccnt 0" ::: "memory");
    __syncthreads();
    compute(As[kt & 1], Bs[kt & 1]);
#else
    // Fallback: synchronous loads staged through VGPRs, single buffer.
    uint4* sA = (uint4*)ldsA[0];
    uint4* sB = (uint4*)ldsB[0];
    for (int kt = 0; kt < NK; ++kt) {
        const uint4* pA = (const uint4*)(gA + (size_t)kt * BK);
        const uint4* pW = (const uint4*)(gW + (size_t)kt * BK);
        uint4 a0 = pA[0], a1 = pA[1];
        uint4 w0 = pW[0], w1 = pW[1];
        __syncthreads();
        sA[0] = a0; sA[1] = a1;
        sB[0] = w0; sB[1] = w1;
        __syncthreads();
        compute(As[0], Bs[0]);
    }
#endif

    // epilogue: C layout -> VGPR r, lane: M = r + 8*hl, N = l15
    #pragma unroll
    for (int sn = 0; sn < 2; ++sn) {
        int n = blockIdx.y * BN + waveN * 32 + sn * 16 + l15;
        if (n < NCLS) {
            float bv = bias[n];
            #pragma unroll
            for (int sm = 0; sm < 4; ++sm) {
                int m0 = blockIdx.x * BM + waveM * 64 + sm * 16 + hl * 8;
                #pragma unroll
                for (int r = 0; r < 8; ++r)
                    out[(size_t)(m0 + r) * NCLS + n] = acc[sm][sn][r] + bv;
            }
        }
    }
}

// ---------------------------------------------------------------------------
// Row softmax over 1000 logits, in place on d_out.
// ---------------------------------------------------------------------------
__global__ __launch_bounds__(128) void softmax_kernel(float* __restrict__ logits)
{
    __shared__ float red[128];
    float* p = logits + (size_t)blockIdx.x * NCLS;
    int t = threadIdx.x;

    float mx = -3.4e38f;
    for (int i = t; i < NCLS; i += 128) mx = fmaxf(mx, p[i]);
    red[t] = mx;
    __syncthreads();
    for (int s = 64; s > 0; s >>= 1) {
        if (t < s) red[t] = fmaxf(red[t], red[t + s]);
        __syncthreads();
    }
    float m = red[0];
    __syncthreads();

    float sum = 0.0f;
    for (int i = t; i < NCLS; i += 128) {
        float e = expf(p[i] - m);
        p[i] = e;
        sum += e;
    }
    red[t] = sum;
    __syncthreads();
    for (int s = 64; s > 0; s >>= 1) {
        if (t < s) red[t] += red[t + s];
        __syncthreads();
    }
    float inv = 1.0f / red[0];
    for (int i = t; i < NCLS; i += 128) p[i] *= inv;
}

// ---------------------------------------------------------------------------
extern "C" void kernel_launch(void* const* d_in, const int* in_sizes, int n_in,
                              void* d_out, int out_size, void* d_ws, size_t ws_size,
                              hipStream_t stream)
{
    const float* x   = (const float*)d_in[0];
    const float* K1  = (const float*)d_in[1];
    const float* K2  = (const float*)d_in[2];
    const float* K3  = (const float*)d_in[3];
    const float* fcw = (const float*)d_in[4];
    const float* fcb = (const float*)d_in[5];
    float* out = (float*)d_out;

    // workspace layout (all offsets 256B aligned)
    char* ws = (char*)d_ws;
    const size_t C1 = (size_t)BATCH * 3 * HW;             // 33,177,600 f32
    const size_t C2 = (size_t)BATCH * 6 * HW;             // 66,355,200 f32
    float*  buf1 = (float*)ws;
    float*  buf2 = (float*)(ws + C1 * 4);
    __bf16* Abf  = (__bf16*)(ws + C1 * 4 + C2 * 4);
    __bf16* Wbf  = (__bf16*)(ws + C1 * 4 + C2 * 4 + (size_t)BATCH * KP * 2);

    conv1_kernel<<<(BATCH * 3 * HW) / 256, 256, 0, stream>>>(x, K1, buf1);
    conv2_kernel<<<(BATCH * 6 * HW) / 256, 256, 0, stream>>>(buf1, K2, buf2);
    conv3_kernel<<<(BATCH * 12 * HW) / 256, 256, 0, stream>>>(buf2, K3, Abf);
    apad_kernel<<<(BATCH * 16) / 256, 256, 0, stream>>>(Abf);
    wcvt_kernel<<<(int)(((size_t)NP * KP) / 1024), 256, 0, stream>>>(fcw, Wbf);

    dim3 grid(BATCH / BM, NP / BN);                       // (32, 8)
    fc_gemm_kernel<<<grid, 256, 0, stream>>>(Abf, Wbf, fcb, out);

    softmax_kernel<<<BATCH, 128, 0, stream>>>(out);
}
